// Model_81226421502266
// MI455X (gfx1250) — matrix operand order repaired and checked by reading, not compiled, not run on hardware
//
#include <hip/hip_runtime.h>
#include <stdint.h>

typedef __attribute__((ext_vector_type(16))) _Float16 v16h;
typedef __attribute__((ext_vector_type(8)))  _Float16 v8h;
typedef __attribute__((ext_vector_type(8)))  float    v8f;

#define B_SZ 1024
#define T_SZ 2048
#define I_SZ 16
#define H_SZ 128
#define G_SZ 512   // 4*H
#define MTILE 16

// ~185 KB LDS: W_ih staged f16 in LDS; W_hh/W_in fragments live in VGPRs.
struct __align__(32) Smem {
  _Float16 Wih[G_SZ * H_SZ];      // 128 KB, [n][k] row-major
  float    gates[MTILE * G_SZ];   // 32 KB f32 gate pre-activations
  float    cst[MTILE * H_SZ];     // 8 KB cell state (f32)
  _Float16 hst[MTILE * H_SZ];     // 4 KB hidden state (f16 WMMA A operand)
  _Float16 xp[MTILE * H_SZ];      // 4 KB relu'd input projection (f16)
  float    xt[2][MTILE * 32];     // 4 KB double-buffered x_t (f32, K padded)
  float    Wout[2 * H_SZ];        // 1 KB
  float    bout[2];
};

// A-fragment c (16x32 f16 slab K=c*32..c*32+31) from one per-lane base:
// p_lane = base + (lane&15)*128 + (lane>>4)*16 ; all offsets are immediates.
__device__ inline v16h afrag(const _Float16* p_lane, int c) {
  v8h lo = *(const v8h*)(p_lane + c * 32);
  v8h hi = *(const v8h*)(p_lane + c * 32 + 8);
  return __builtin_shufflevector(lo, hi, 0,1,2,3,4,5,6,7,8,9,10,11,12,13,14,15);
}

// B-fragment c from LDS W[n][k]: p_lane = W + ncol*128 + (lane>>4)*16.
__device__ inline v16h bfrag(const _Float16* p_lane, int c) {
  return *(const v16h*)(p_lane + c * 32);
}

// A-fragment from f32 LDS x tile: p_lane = xt + (lane&15)*32 + (lane>>4)*16.
__device__ inline v16h afrag_f32(const float* p_lane) {
  float4 a0 = *(const float4*)(p_lane + 0);
  float4 a1 = *(const float4*)(p_lane + 4);
  float4 a2 = *(const float4*)(p_lane + 8);
  float4 a3 = *(const float4*)(p_lane + 12);
  v16h r;
  r[0]=(_Float16)a0.x;  r[1]=(_Float16)a0.y;  r[2]=(_Float16)a0.z;  r[3]=(_Float16)a0.w;
  r[4]=(_Float16)a1.x;  r[5]=(_Float16)a1.y;  r[6]=(_Float16)a1.z;  r[7]=(_Float16)a1.w;
  r[8]=(_Float16)a2.x;  r[9]=(_Float16)a2.y;  r[10]=(_Float16)a2.z; r[11]=(_Float16)a2.w;
  r[12]=(_Float16)a3.x; r[13]=(_Float16)a3.y; r[14]=(_Float16)a3.z; r[15]=(_Float16)a3.w;
  return r;
}

// B-fragment loaded once from global f32 W stored [n][k].
__device__ inline v16h bfrag_g(const float* w, int stride, int n, int k) {
  const float* p = w + n * stride + k;
  float4 q0 = *(const float4*)(p + 0);
  float4 q1 = *(const float4*)(p + 4);
  float4 q2 = *(const float4*)(p + 8);
  float4 q3 = *(const float4*)(p + 12);
  v16h r;
  r[0]=(_Float16)q0.x;  r[1]=(_Float16)q0.y;  r[2]=(_Float16)q0.z;  r[3]=(_Float16)q0.w;
  r[4]=(_Float16)q1.x;  r[5]=(_Float16)q1.y;  r[6]=(_Float16)q1.z;  r[7]=(_Float16)q1.w;
  r[8]=(_Float16)q2.x;  r[9]=(_Float16)q2.y;  r[10]=(_Float16)q2.z; r[11]=(_Float16)q2.w;
  r[12]=(_Float16)q3.x; r[13]=(_Float16)q3.y; r[14]=(_Float16)q3.z; r[15]=(_Float16)q3.w;
  return r;
}

__device__ inline float sigmoidf_(float v) { return 1.0f / (1.0f + __expf(-v)); }
__device__ inline float tanhf_(float v) {   // exp on TRANS pipe
  float e = __expf(-2.0f * v);
  return (1.0f - e) / (1.0f + e);
}

extern "C" __global__ __launch_bounds__(1024, 1)
void lstm_fused(const float* __restrict__ x,
                const float* __restrict__ W_in, const float* __restrict__ b_in,
                const float* __restrict__ W_ih, const float* __restrict__ W_hh,
                const float* __restrict__ b_ih, const float* __restrict__ b_hh,
                const float* __restrict__ W_out, const float* __restrict__ b_out,
                float* __restrict__ out)
{
  __shared__ Smem sm;
  const int tid  = threadIdx.x;
  const int lane = tid & 31;
  const int wave = tid >> 5;
  const int b0   = blockIdx.x * MTILE;
  const int n0   = wave << 4;            // this wave's 16-wide gate column tile
  const int ncol = n0 + (lane & 15);
  const int hi16 = lane >> 4;            // 0 or 1

  // ---- one-time: W_hh / W_in B-fragments into VGPRs ----
  v16h Bhh[4];
#pragma unroll
  for (int c = 0; c < 4; ++c)
    Bhh[c] = bfrag_g(W_hh, H_SZ, ncol, c * 32 + (hi16 << 4));
  v16h Bin;
  if (wave < 8) {
    if (lane < 16) {
      Bin = bfrag_g(W_in, I_SZ, ncol, 0);        // K=0..15 valid
    } else {
#pragma unroll
      for (int j = 0; j < 16; ++j) Bin[j] = (_Float16)0.0f;  // zero pad
    }
  }
  const float bias_g = b_ih[ncol] + b_hh[ncol];
  float bias_p = 0.0f;
  if (wave < 8) bias_p = b_in[ncol];

  // ---- one-time: LDS init (W_ih f32->f16 packed, state zero, x_0 preload) ----
  for (int i = tid; i < (G_SZ * H_SZ) / 2; i += 1024) {
    float2 w = *(const float2*)(W_ih + 2 * i);
    sm.Wih[2 * i]     = (_Float16)w.x;
    sm.Wih[2 * i + 1] = (_Float16)w.y;
  }
  if (tid < 2 * H_SZ) sm.Wout[tid] = W_out[tid];
  if (tid < 2)        sm.bout[tid] = b_out[tid];
  for (int i = tid; i < MTILE * H_SZ; i += 1024) { sm.cst[i] = 0.0f; sm.hst[i] = (_Float16)0.0f; }
  if (tid < 512) {               // zero K=16..31 pad of both x buffers
    int bu = tid >> 8, m = (tid >> 4) & 15, i = 16 + (tid & 15);
    sm.xt[bu][m * 32 + i] = 0.0f;
  }
  if (tid < MTILE * I_SZ) {      // preload x_0
    int m = tid >> 4, i = tid & 15;
    sm.xt[0][m * 32 + i] = x[((size_t)(b0 + m) * T_SZ + 0) * I_SZ + i];
  }
  __syncthreads();

  // ---- per-lane base pointers (single VGPR each; row offsets are immediates) ----
  const _Float16* pA_h  = sm.hst + (lane & 15) * H_SZ + hi16 * 16;
  const _Float16* pA_xp = sm.xp  + (lane & 15) * H_SZ + hi16 * 16;
  const _Float16* pB_ih = sm.Wih + ncol * H_SZ + hi16 * 16;
  _Float16*       pS_xp = sm.xp    + (hi16 << 3) * H_SZ + ncol;
  float*          pS_g  = sm.gates + (hi16 << 3) * G_SZ + ncol;

  for (int t = 0; t < T_SZ; ++t) {
    const int buf = t & 1;

    // 0) async-copy x_{t+1} into the other buffer (ASYNCcnt, overlaps the step)
    if (t + 1 < T_SZ && tid < MTILE * I_SZ) {
      int m = tid >> 4, i = tid & 15;
      uint32_t ldst = (uint32_t)(uintptr_t)&sm.xt[buf ^ 1][m * 32 + i];
      const float* src = &x[((size_t)(b0 + m) * T_SZ + (t + 1)) * I_SZ + i];
      asm volatile("global_load_async_to_lds_b32 %0, %1, off"
                   :: "v"(ldst), "v"((uint64_t)(uintptr_t)src) : "memory");
    }

    // 1) waves 0..7: xp = relu(x_t @ W_in^T + b_in)  (one WMMA, write f16)
    if (wave < 8) {
      v8f accp = {bias_p, bias_p, bias_p, bias_p, bias_p, bias_p, bias_p, bias_p};
      v16h a = afrag_f32(sm.xt[buf] + (lane & 15) * 32 + hi16 * 16);
      accp = __builtin_amdgcn_wmma_f32_16x16x32_f16(false, a, false, Bin, (short)0, accp, false, false);
#pragma unroll
      for (int r = 0; r < 8; ++r) {
        float v = accp[r]; v = v > 0.0f ? v : 0.0f;
        pS_xp[r * H_SZ] = (_Float16)v;
      }
    }

    // 2a) ALL waves: h @ Whh^T (h already synced by last iteration's barrier)
    v8f acc = {bias_g, bias_g, bias_g, bias_g, bias_g, bias_g, bias_g, bias_g};
#pragma unroll
    for (int c = 0; c < 4; ++c) {
      v16h a = afrag(pA_h, c);
      acc = __builtin_amdgcn_wmma_f32_16x16x32_f16(false, a, false, Bhh[c], (short)0, acc, false, false);
    }
    __syncthreads();   // xp now visible

    // 2b) ALL waves: += xp @ Wih^T (B from LDS), then publish gates
#pragma unroll
    for (int c = 0; c < 4; ++c) {
      v16h a = afrag(pA_xp, c);
      v16h b = bfrag(pB_ih, c);
      acc = __builtin_amdgcn_wmma_f32_16x16x32_f16(false, a, false, b, (short)0, acc, false, false);
    }
#pragma unroll
    for (int r = 0; r < 8; ++r) pS_g[r * G_SZ] = acc[r];
    __syncthreads();   // gates visible

    // 3) LSTM cell: 2048 elements, 2 per thread (exp on TRANS pipe)
#pragma unroll
    for (int q = 0; q < 2; ++q) {
      int e = tid * 2 + q;
      int m = e >> 7, j = e & 127;
      float ig = sigmoidf_(sm.gates[m * G_SZ + j]);
      float fg = sigmoidf_(sm.gates[m * G_SZ + 128 + j]);
      float gg = tanhf_(sm.gates[m * G_SZ + 256 + j]);
      float og = sigmoidf_(sm.gates[m * G_SZ + 384 + j]);
      float c  = fg * sm.cst[m * H_SZ + j] + ig * gg;
      float h  = og * tanhf_(c);
      sm.cst[m * H_SZ + j] = c;
      sm.hst[m * H_SZ + j] = (_Float16)h;
    }
    __syncthreads();   // h visible

    // 4) output projection: wave w handles batch row m=w; wave32 shuffle reduce
    if (wave < MTILE) {
      float s0 = 0.0f, s1 = 0.0f;
#pragma unroll
      for (int q = 0; q < 4; ++q) {
        int j = lane * 4 + q;
        float hv = (float)sm.hst[wave * H_SZ + j];
        s0 += hv * sm.Wout[j];
        s1 += hv * sm.Wout[H_SZ + j];
      }
#pragma unroll
      for (int off = 16; off > 0; off >>= 1) {
        s0 += __shfl_xor(s0, off, 32);
        s1 += __shfl_xor(s1, off, 32);
      }
      if (lane == 0) {
        size_t base = (size_t)(b0 + wave) * T_SZ + t;
        out[base] = s0 + sm.bout[0];
        float lv = s1 + sm.bout[1];
        out[(size_t)B_SZ * T_SZ + base] = lv > 0.0f ? lv : 0.0f;
      }
    }

    // 5) next x tile must be resident before anyone reads it
    asm volatile("s_wait_asynccnt 0x0" ::: "memory");
    __syncthreads();
  }
}

extern "C" void kernel_launch(void* const* d_in, const int* in_sizes, int n_in,
                              void* d_out, int out_size, void* d_ws, size_t ws_size,
                              hipStream_t stream) {
  const float* x     = (const float*)d_in[0];
  const float* W_in  = (const float*)d_in[1];
  const float* b_in  = (const float*)d_in[2];
  const float* W_ih  = (const float*)d_in[3];
  const float* W_hh  = (const float*)d_in[4];
  const float* b_ih  = (const float*)d_in[5];
  const float* b_hh  = (const float*)d_in[6];
  const float* W_out = (const float*)d_in[7];
  const float* b_out = (const float*)d_in[8];
  float* out = (float*)d_out;

  dim3 grid(B_SZ / MTILE);   // 64 workgroups, one per 16-row batch tile
  dim3 block(1024);          // 32 wave32s: one per 16-wide gate column tile
  hipLaunchKernelGGL(lstm_fused, grid, block, 0, stream,
                     x, W_in, b_in, W_ih, W_hh, b_ih, b_hh, W_out, b_out, out);
}
